// QuantumPredictorLSTM_45037027066357
// MI455X (gfx1250) — matrix operand order repaired and checked by reading, not compiled, not run on hardware
//
#include <hip/hip_runtime.h>
#include <hip/hip_bf16.h>

typedef _Float16 v16h __attribute__((ext_vector_type(16)));
typedef _Float16 v8h  __attribute__((ext_vector_type(8)));
typedef float    v8f  __attribute__((ext_vector_type(8)));

#define Tn   64
#define Hn   32
#define GATES 128
#define ROWS_PER_BLOCK 128
#define ROWS_PER_WAVE  16
#define XPAD 136   // pad so 16B vector loads stay aligned & banks spread

// ---- activations: prefer CDNA5 hardware v_tanh_f32 ----
#if __has_builtin(__builtin_amdgcn_tanhf)
__device__ __forceinline__ float ftanh(float x) { return __builtin_amdgcn_tanhf(x); }
#define HAVE_HW_TANH 1
#elif __has_builtin(__builtin_amdgcn_tanh_f32)
__device__ __forceinline__ float ftanh(float x) { return __builtin_amdgcn_tanh_f32(x); }
#define HAVE_HW_TANH 1
#else
__device__ __forceinline__ float ftanh(float x) {
    float e = __builtin_amdgcn_exp2f(2.88539008f * x);
    return 1.0f - 2.0f * __builtin_amdgcn_rcpf(e + 1.0f);
}
#define HAVE_HW_TANH 0
#endif

__device__ __forceinline__ float fsig(float x) {
#if HAVE_HW_TANH
    // sigmoid(x) = 0.5*tanh(x/2) + 0.5  -> single trans op
    return __builtin_fmaf(0.5f, ftanh(0.5f * x), 0.5f);
#else
    return __builtin_amdgcn_rcpf(1.0f + __builtin_amdgcn_exp2f(-1.44269504f * x));
#endif
}

__global__ __launch_bounds__(256, 4)
void lstm2_wmma_kernel(const float* __restrict__ x,
                       const float* __restrict__ W_ih0,
                       const float* __restrict__ W_hh0,
                       const float* __restrict__ b_ih0,
                       const float* __restrict__ b_hh0,
                       const float* __restrict__ W_ih1,
                       const float* __restrict__ W_hh1,
                       const float* __restrict__ b_ih1,
                       const float* __restrict__ b_hh1,
                       const float* __restrict__ W_fc,
                       const float* __restrict__ b_fc,
                       float* __restrict__ out)
{
    __shared__ _Float16 sx[Tn * XPAD];           // x tile, transposed [t][row]
    __shared__ _Float16 sW0[GATES * Hn];         // W_hh0 (f16)  [gate][k]
    __shared__ _Float16 sWi1[GATES * Hn];        // W_ih1 (f16)
    __shared__ _Float16 sWh1[GATES * Hn];        // W_hh1 (f16)
    __shared__ float    sb1[GATES], sb2[GATES], swih0[GATES];
    __shared__ float    swfc[Hn];
    __shared__ _Float16 h1buf[8][ROWS_PER_WAVE * Hn]; // per-wave transpose scratch
    __shared__ _Float16 h2buf[8][ROWS_PER_WAVE * Hn];

    const int tid = threadIdx.x;
    const int blockBase = blockIdx.x * ROWS_PER_BLOCK;

    // ---- stage inputs (coalesced global reads) ----
    for (int i = tid; i < ROWS_PER_BLOCK * Tn; i += 256) {
        int bl = i >> 6, t = i & 63;
        sx[t * XPAD + bl] = (_Float16)x[blockBase * Tn + i];
    }
    for (int i = tid; i < GATES * Hn; i += 256) {
        sW0[i]  = (_Float16)W_hh0[i];
        sWi1[i] = (_Float16)W_ih1[i];
        sWh1[i] = (_Float16)W_hh1[i];
    }
    if (tid < GATES) {
        sb1[tid]   = b_ih0[tid] + b_hh0[tid];
        sb2[tid]   = b_ih1[tid] + b_hh1[tid];
        swih0[tid] = W_ih0[tid];
    }
    if (tid < Hn) swfc[tid] = W_fc[tid];
    __syncthreads();

    const int wave = tid >> 5;
    const int lane = tid & 31;
    const int ln   = lane & 15;   // N (gate col) / M (row) index
    const int hi   = lane >> 4;   // lane half selects K / M sub-range
    const int waveRow = wave * ROWS_PER_WAVE;

    // per-lane gate-tile constants (gate = p*16 + ln)
    float wih0v[8], b1v[8], b2v[8];
#pragma unroll
    for (int p = 0; p < 8; ++p) {
        int g = p * 16 + ln;
        wih0v[p] = swih0[g]; b1v[p] = sb1[g]; b2v[p] = sb2[g];
    }

    v16h a_h1 = {};                 // A fragments (h state, f16, WMMA A layout)
    v16h a_h2 = {};
    v8f c1a = {}, c1b = {};         // cell state, C/D layout (n 0..15 / 16..31)
    v8f c2a = {}, c2b = {};

    _Float16* myh1 = h1buf[wave];
    _Float16* myh2 = h2buf[wave];

    for (int t = 0; t < Tn; ++t) {
        // opaque zero: forces W_ih1/W_hh1 fragment loads to stay inside the
        // loop (re-read from LDS each step) instead of being hoisted into
        // ~128 extra live VGPRs. W_hh0 fragments remain register-resident.
        unsigned wofs = 0;
        asm volatile("" : "+v"(wofs));

        // x values for this lane's 8 rows (M = hi*8 + r)
        v8h xh = *(const v8h*)&sx[t * XPAD + waveRow + hi * 8];
        float xr[8];
#pragma unroll
        for (int r = 0; r < 8; ++r) xr[r] = (float)xh[r];

        // ---- layer 1: g = (x*W_ih0 + b) + h1 @ W_hh0^T ----
        // hidden-set s uses gate tiles {s, 2+s, 4+s, 6+s} = i,f,g,o
#pragma unroll
        for (int s = 0; s < 2; ++s) {
            v8f g[4];
#pragma unroll
            for (int q = 0; q < 4; ++q) {
                const int p = q * 2 + s;
                v8f c;
#pragma unroll
                for (int r = 0; r < 8; ++r) c[r] = xr[r] * wih0v[p] + b1v[p];
                v16h bf = *(const v16h*)&sW0[(p * 16 + ln) * Hn + hi * 16];
                g[q] = __builtin_amdgcn_wmma_f32_16x16x32_f16(
                           false, a_h1, false, bf, (short)0, c, false, false);
            }
            v8f& cs = s ? c1b : c1a;
#pragma unroll
            for (int r = 0; r < 8; ++r) {
                float cn = fsig(g[1][r]) * cs[r] + fsig(g[0][r]) * ftanh(g[2][r]);
                cs[r] = cn;
                float hn = fsig(g[3][r]) * ftanh(cn);
                myh1[(hi * 8 + r) * Hn + s * 16 + ln] = (_Float16)hn;
            }
        }
        // rebuild A fragment for h1 (row M=ln; K = hi*16..+15)
        {
            v8h lo = *(const v8h*)&myh1[ln * Hn + hi * 16];
            v8h hh = *(const v8h*)&myh1[ln * Hn + hi * 16 + 8];
            a_h1 = __builtin_shufflevector(lo, hh, 0,1,2,3,4,5,6,7,8,9,10,11,12,13,14,15);
        }

        // ---- layer 2: g = b + h1 @ W_ih1^T + h2 @ W_hh1^T ----
#pragma unroll
        for (int s = 0; s < 2; ++s) {
            v8f g[4];
#pragma unroll
            for (int q = 0; q < 4; ++q) {
                const int p = q * 2 + s;
                v8f c;
#pragma unroll
                for (int r = 0; r < 8; ++r) c[r] = b2v[p];
                v16h bi = *(const v16h*)&sWi1[(p * 16 + ln) * Hn + hi * 16 + wofs];
                c = __builtin_amdgcn_wmma_f32_16x16x32_f16(
                        false, a_h1, false, bi, (short)0, c, false, false);
                v16h bh = *(const v16h*)&sWh1[(p * 16 + ln) * Hn + hi * 16 + wofs];
                g[q] = __builtin_amdgcn_wmma_f32_16x16x32_f16(
                        false, a_h2, false, bh, (short)0, c, false, false);
            }
            v8f& cs = s ? c2b : c2a;
#pragma unroll
            for (int r = 0; r < 8; ++r) {
                float cn = fsig(g[1][r]) * cs[r] + fsig(g[0][r]) * ftanh(g[2][r]);
                cs[r] = cn;
                float hn = fsig(g[3][r]) * ftanh(cn);
                myh2[(hi * 8 + r) * Hn + s * 16 + ln] = (_Float16)hn;
            }
        }
        {
            v8h lo = *(const v8h*)&myh2[ln * Hn + hi * 16];
            v8h hh = *(const v8h*)&myh2[ln * Hn + hi * 16 + 8];
            a_h2 = __builtin_shufflevector(lo, hh, 0,1,2,3,4,5,6,7,8,9,10,11,12,13,14,15);
        }
    }

    // ---- final FC + sigmoid: out[b] = sig(h2_last . wfc + bfc) ----
    if (hi == 0) {
        float acc = b_fc[0];
#pragma unroll
        for (int k = 0; k < Hn; ++k)
            acc += (float)myh2[ln * Hn + k] * swfc[k];
        out[blockBase + waveRow + ln] = fsig(acc);
    }
}

extern "C" void kernel_launch(void* const* d_in, const int* in_sizes, int n_in,
                              void* d_out, int out_size, void* d_ws, size_t ws_size,
                              hipStream_t stream) {
    const float* x     = (const float*)d_in[0];
    const float* W_ih0 = (const float*)d_in[1];
    const float* W_hh0 = (const float*)d_in[2];
    const float* b_ih0 = (const float*)d_in[3];
    const float* b_hh0 = (const float*)d_in[4];
    const float* W_ih1 = (const float*)d_in[5];
    const float* W_hh1 = (const float*)d_in[6];
    const float* b_ih1 = (const float*)d_in[7];
    const float* b_hh1 = (const float*)d_in[8];
    const float* W_fc  = (const float*)d_in[9];
    const float* b_fc  = (const float*)d_in[10];
    float* out = (float*)d_out;

    int B = in_sizes[0] / Tn;                 // 65536
    int grid = B / ROWS_PER_BLOCK;            // 512 blocks of 256 threads (8 waves)
    lstm2_wmma_kernel<<<grid, 256, 0, stream>>>(
        x, W_ih0, W_hh0, b_ih0, b_hh0, W_ih1, W_hh1, b_ih1, b_hh1, W_fc, b_fc, out);
}
